// TransformerVisionLayer_63754494542010
// MI455X (gfx1250) — matrix operand order repaired and checked by reading, not compiled, not run on hardware
//
#include <hip/hip_runtime.h>

#define DEV __device__ __forceinline__

typedef __bf16 v8bf  __attribute__((ext_vector_type(8)));
typedef __bf16 v16bf __attribute__((ext_vector_type(16)));
typedef float  v8f   __attribute__((ext_vector_type(8)));
typedef int    v4i   __attribute__((ext_vector_type(4)));

constexpr int kB = 8, kN = 4096, kH = 768, kNH = 12, kDH = 64, kFF = 3072;
constexpr long long kBN = (long long)kB * kN;   // 32768 rows

constexpr int EPI_BF16 = 1, EPI_F32 = 2, EPI_BIAS = 4, EPI_RELU = 8, EPI_HEADS = 16;

#if defined(__has_builtin)
#if __has_builtin(__builtin_amdgcn_global_load_async_to_lds_b128) && \
    __has_builtin(__builtin_amdgcn_s_wait_asynccnt)
#define HAVE_ASYNC_LDS 1
#endif
#endif

DEV unsigned short f2bf(float f) {
  unsigned u = __float_as_uint(f);
  u += 0x7fffu + ((u >> 16) & 1u);          // round-to-nearest-even
  return (unsigned short)(u >> 16);
}
DEV float bf2f(unsigned short h) { return __uint_as_float(((unsigned)h) << 16); }

DEV v16bf cat16(v8bf lo, v8bf hi) {
  return __builtin_shufflevector(lo, hi, 0,1,2,3,4,5,6,7,8,9,10,11,12,13,14,15);
}

#ifdef HAVE_ASYNC_LDS
typedef __attribute__((address_space(1))) v4i* gptr_v4i;
typedef __attribute__((address_space(3))) v4i* lptr_v4i;
DEV void async_copy16(const unsigned short* g, unsigned short* l) {
  __builtin_amdgcn_global_load_async_to_lds_b128((gptr_v4i)g, (lptr_v4i)l, 0, 0);
}
#endif

// ---------------------------------------------------------------- cast f32->bf16
__global__ __launch_bounds__(256) void cast_f32_bf16(const float* __restrict__ s,
                                                     unsigned short* __restrict__ d, int n) {
  int i = blockIdx.x * 256 + threadIdx.x;
  if (i < n) d[i] = f2bf(s[i]);
}

// ---------------------------------------------------------------- k += key_pe (broadcast over batch & heads)
__global__ __launch_bounds__(256) void add_pe_kernel(unsigned short* __restrict__ k,
                                                     const float* __restrict__ pe) {
  long long i = (long long)blockIdx.x * 256 + threadIdx.x;   // exactly BN*H threads
  int h = (int)(i % kH);
  int n = (int)((i / kH) % kN);
  k[i] = f2bf(bf2f(k[i]) + pe[n * kDH + (h & 63)]);
}

// ---------------------------------------------------------------- generic bf16 GEMM: C = A (M x K, lda) * Bw^T (Nc x K, ldb)
// block: 256 threads = 8 waves (4 x 2). Block tile 128 x (32*TN), BK = 32,
// double-buffered LDS (async global->LDS when available), 1 barrier / K-step.
template <int TN>   // wave column tiles: TN=4 -> 128x128 block, TN=2 -> 128x64
__global__ __launch_bounds__(256) void gemm_bf16_kernel(
    const unsigned short* __restrict__ A, int lda,
    const unsigned short* __restrict__ Bw, int ldb,
    unsigned short* __restrict__ Cb, float* __restrict__ Cf, int ldc,
    const float* __restrict__ bias, int K, int flags)
{
  constexpr int BNC = 32 * TN;
  __shared__ __align__(16) unsigned short As[2][128][40];   // 32 cols + pad (16B rows)
  __shared__ __align__(16) unsigned short Bs[2][BNC][40];

  if (flags & EPI_HEADS) {            // grid.z = B*NH; A/C are per-head slices, Bw batched
    int z = blockIdx.z;
    size_t hoff = (size_t)(z / kNH) * (size_t)kN * kH + (size_t)(z % kNH) * kDH;
    A += hoff;
    if (Cb) Cb += hoff;
    if (Cf) Cf += hoff;
    Bw += (size_t)z * kDH * kDH;
  }
  const int t = threadIdx.x;
  const int lane = t & 31, wave = t >> 5;
  const int wm = wave & 3, wn = wave >> 2;
  const int lm = lane & 15, lk = lane >> 4;
  const size_t rowBase = (size_t)blockIdx.y * 128;
  const size_t colBase = (size_t)blockIdx.x * BNC;

  v8f acc[2][TN] = {};

#ifndef HAVE_ASYNC_LDS
  uint4 regA[2];
  uint4 regB[TN / 2];
#endif

  // Stage tile for k0 into LDS buffer `buf`: A is 128x32 (512 x 16B), B is BNCx32.
  auto stage_issue = [&](int k0, int buf) {
#pragma unroll
    for (int s = 0; s < 2; s++) {
      int slot = t + s * 256;
      int r = slot >> 2, c = (slot & 3) << 3;
      const unsigned short* g = A + (rowBase + r) * (size_t)lda + (k0 + c);
#ifdef HAVE_ASYNC_LDS
      async_copy16(g, &As[buf][r][c]);
#else
      regA[s] = *(const uint4*)g;
#endif
    }
#pragma unroll
    for (int s = 0; s < TN / 2; s++) {
      int slot = t + s * 256;
      int r = slot >> 2, c = (slot & 3) << 3;
      const unsigned short* g = Bw + (colBase + r) * (size_t)ldb + (k0 + c);
#ifdef HAVE_ASYNC_LDS
      async_copy16(g, &Bs[buf][r][c]);
#else
      regB[s] = *(const uint4*)g;
#endif
    }
  };
  auto stage_commit = [&](int buf) {
#ifdef HAVE_ASYNC_LDS
    __builtin_amdgcn_s_wait_asynccnt(0);
#else
#pragma unroll
    for (int s = 0; s < 2; s++) {
      int slot = t + s * 256;
      int r = slot >> 2, c = (slot & 3) << 3;
      *(uint4*)&As[buf][r][c] = regA[s];
    }
#pragma unroll
    for (int s = 0; s < TN / 2; s++) {
      int slot = t + s * 256;
      int r = slot >> 2, c = (slot & 3) << 3;
      *(uint4*)&Bs[buf][r][c] = regB[s];
    }
#endif
  };

  stage_issue(0, 0);
  stage_commit(0);
  __syncthreads();

  int cur = 0;
  for (int k0 = 0; k0 < K; k0 += 32, cur ^= 1) {
    const bool more = (k0 + 32) < K;
    if (more) stage_issue(k0 + 32, cur ^ 1);   // overlap next-tile fetch with compute

    v16bf af[2];
#pragma unroll
    for (int tm = 0; tm < 2; tm++) {
      int r = wm * 32 + tm * 16 + lm;
      af[tm] = cat16(*(const v8bf*)&As[cur][r][lk * 8],
                     *(const v8bf*)&As[cur][r][16 + lk * 8]);
    }
#pragma unroll
    for (int tn = 0; tn < TN; tn++) {
      int r = wn * (16 * TN) + tn * 16 + lm;
      v16bf bfr = cat16(*(const v8bf*)&Bs[cur][r][lk * 8],
                        *(const v8bf*)&Bs[cur][r][16 + lk * 8]);
#pragma unroll
      for (int tm = 0; tm < 2; tm++)
        acc[tm][tn] = __builtin_amdgcn_wmma_f32_16x16x32_bf16(
            false, af[tm], false, bfr, (short)0, acc[tm][tn], false, false);
    }
    if (more) stage_commit(cur ^ 1);
    __syncthreads();
  }

#pragma unroll
  for (int tm = 0; tm < 2; tm++)
#pragma unroll
    for (int tn = 0; tn < TN; tn++) {
      size_t gcol = colBase + wn * (16 * TN) + tn * 16 + lm;
      float bv = (flags & EPI_BIAS) ? bias[gcol] : 0.0f;
#pragma unroll
      for (int r = 0; r < 8; r++) {
        size_t grow = rowBase + wm * 32 + tm * 16 + lk * 8 + r;
        float val = acc[tm][tn][r] + bv;
        if (flags & EPI_RELU) val = fmaxf(val, 0.0f);
        if (flags & EPI_BF16) Cb[grow * (size_t)ldc + gcol] = f2bf(val);
        else                  Cf[grow * (size_t)ldc + gcol] = val;
      }
    }
}

// ---------------------------------------------------------------- attention logits: L[z,d,e] = sum_n q[z,n,d] * k'[z,n,e]
// One block per (b,head). Both operands transposed-staged into LDS so fragment reads are K-contiguous.
__global__ __launch_bounds__(256) void attn_logits_kernel(
    const unsigned short* __restrict__ q, const unsigned short* __restrict__ kp,
    float* __restrict__ logits)
{
  __shared__ __align__(16) unsigned short qT[64][40];   // [d][n(0..31)]
  __shared__ __align__(16) unsigned short kT[64][40];   // [e][n(0..31)]

  const int z = blockIdx.x;
  const size_t hoff = (size_t)(z / kNH) * (size_t)kN * kH + (size_t)(z % kNH) * kDH;
  const unsigned short* qh = q + hoff;
  const unsigned short* kh = kp + hoff;

  const int t = threadIdx.x;
  const int lane = t & 31, wave = t >> 5;
  const int lm = lane & 15, lk = lane >> 4;

  v8f acc[2] = {};

  for (int n0 = 0; n0 < kN; n0 += 32) {
    int n = t >> 3, d8 = (t & 7) << 3;
    uint4 qv = *(const uint4*)(qh + (size_t)(n0 + n) * kH + d8);
    uint4 kv = *(const uint4*)(kh + (size_t)(n0 + n) * kH + d8);
    const unsigned short* qs = (const unsigned short*)&qv;
    const unsigned short* ks = (const unsigned short*)&kv;
#pragma unroll
    for (int j = 0; j < 8; j++) { qT[d8 + j][n] = qs[j]; kT[d8 + j][n] = ks[j]; }
    __syncthreads();

#pragma unroll
    for (int i = 0; i < 2; i++) {
      int tile = wave * 2 + i, td = tile >> 2, te = tile & 3;
      int ra = td * 16 + lm, rb = te * 16 + lm;
      v16bf af = cat16(*(const v8bf*)&qT[ra][lk * 8], *(const v8bf*)&qT[ra][16 + lk * 8]);
      v16bf bf_ = cat16(*(const v8bf*)&kT[rb][lk * 8], *(const v8bf*)&kT[rb][16 + lk * 8]);
      acc[i] = __builtin_amdgcn_wmma_f32_16x16x32_bf16(
          false, af, false, bf_, (short)0, acc[i], false, false);
    }
    __syncthreads();
  }

#pragma unroll
  for (int i = 0; i < 2; i++) {
    int tile = wave * 2 + i, td = tile >> 2, te = tile & 3;
#pragma unroll
    for (int r = 0; r < 8; r++) {
      int d = td * 16 + lk * 8 + r, e = te * 16 + lm;
      logits[(size_t)z * 4096 + d * 64 + e] = acc[i][r];
    }
  }
}

// ---------------------------------------------------------------- softmax over e (64), scale 1/sqrt(DH)
__global__ __launch_bounds__(256) void softmax_kernel(const float* __restrict__ logits,
                                                      unsigned short* __restrict__ probs) {
  const int z = blockIdx.x;
  const int lane = threadIdx.x & 31, wave = threadIdx.x >> 5;
  for (int d = wave; d < 64; d += 8) {
    const float* row = logits + (size_t)z * 4096 + d * 64;
    float v0 = row[lane] * 0.125f, v1 = row[lane + 32] * 0.125f;
    float m = fmaxf(v0, v1);
#pragma unroll
    for (int off = 16; off > 0; off >>= 1) m = fmaxf(m, __shfl_xor(m, off, 32));
    float e0 = __expf(v0 - m), e1 = __expf(v1 - m);
    float s = e0 + e1;
#pragma unroll
    for (int off = 16; off > 0; off >>= 1) s += __shfl_xor(s, off, 32);
    float inv = 1.0f / s;
    probs[(size_t)z * 4096 + d * 64 + lane]      = f2bf(e0 * inv);
    probs[(size_t)z * 4096 + d * 64 + lane + 32] = f2bf(e1 * inv);
  }
}

// ---------------------------------------------------------------- residual + LayerNorm (row = 768, block = 256)
DEV float block_reduce_sum(float s, float* red) {
  const int lane = threadIdx.x & 31, wave = threadIdx.x >> 5;
#pragma unroll
  for (int off = 16; off > 0; off >>= 1) s += __shfl_xor(s, off, 32);
  if (lane == 0) red[wave] = s;
  __syncthreads();
  float tot = 0.0f;
#pragma unroll
  for (int w = 0; w < 8; w++) tot += red[w];
  __syncthreads();
  return tot;
}

__global__ __launch_bounds__(256) void ln_residual_kernel(
    const float* __restrict__ xin, const float* __restrict__ delta,
    const float* __restrict__ gamma, const float* __restrict__ beta,
    float* __restrict__ yf, unsigned short* __restrict__ yb)
{
  __shared__ float red[8];
  const size_t row = blockIdx.x;
  const float* xr = xin + row * kH;
  const float* dr = delta + row * kH;
  float v[3]; float s = 0.0f;
#pragma unroll
  for (int i = 0; i < 3; i++) { int c = threadIdx.x + i * 256; v[i] = xr[c] + dr[c]; s += v[i]; }
  float mean = block_reduce_sum(s, red) * (1.0f / 768.0f);
  float vs = 0.0f;
#pragma unroll
  for (int i = 0; i < 3; i++) { float d = v[i] - mean; vs += d * d; }
  float inv = rsqrtf(block_reduce_sum(vs, red) * (1.0f / 768.0f) + 1e-5f);
#pragma unroll
  for (int i = 0; i < 3; i++) {
    int c = threadIdx.x + i * 256;
    float y = (v[i] - mean) * inv * gamma[c] + beta[c];
    yf[row * kH + c] = y;
    if (yb) yb[row * kH + c] = f2bf(y);
  }
}

// ---------------------------------------------------------------- launcher
extern "C" void kernel_launch(void* const* d_in, const int* in_sizes, int n_in,
                              void* d_out, int out_size, void* d_ws, size_t ws_size,
                              hipStream_t stream) {
  const float* x      = (const float*)d_in[0];
  const float* key_pe = (const float*)d_in[1];
  const float* Wq     = (const float*)d_in[2];
  const float* Wk     = (const float*)d_in[3];
  const float* Wv     = (const float*)d_in[4];
  const float* Wo     = (const float*)d_in[5];
  const float* fc1_w  = (const float*)d_in[6];
  const float* fc1_b  = (const float*)d_in[7];
  const float* fc2_w  = (const float*)d_in[8];
  const float* fc2_b  = (const float*)d_in[9];
  const float* ln1_g  = (const float*)d_in[10];
  const float* ln1_b  = (const float*)d_in[11];
  const float* ln3_g  = (const float*)d_in[12];
  const float* ln3_b  = (const float*)d_in[13];

  char* p = (char*)d_ws;
  auto take = [&](size_t bytes) -> char* {
    char* r = p; p += (bytes + 255) & ~(size_t)255; return r;
  };
  const size_t actB = (size_t)kBN * kH * 2;        // bf16 activation [32768,768]
  const size_t actF = (size_t)kBN * kH * 4;        // f32 activation
  unsigned short* xb    = (unsigned short*)take(actB);
  unsigned short* wqb   = (unsigned short*)take((size_t)kH * kH * 2);
  unsigned short* wkb   = (unsigned short*)take((size_t)kH * kH * 2);
  unsigned short* wvb   = (unsigned short*)take((size_t)kH * kH * 2);
  unsigned short* wob   = (unsigned short*)take((size_t)kH * kH * 2);
  unsigned short* fc1wb = (unsigned short*)take((size_t)kFF * kH * 2);
  unsigned short* fc2wb = (unsigned short*)take((size_t)kH * kFF * 2);
  unsigned short* qb    = (unsigned short*)take(actB);
  unsigned short* kb    = (unsigned short*)take(actB);
  unsigned short* vb    = (unsigned short*)take(actB);
  float*          logit = (float*)take((size_t)kB * kNH * kDH * kDH * 4);
  unsigned short* probs = (unsigned short*)take((size_t)kB * kNH * kDH * kDH * 2);
  float*          proj  = (float*)take(actF);       // Wo output; reused for fc2 output
  float*          x1    = (float*)take(actF);       // LN1 output (f32 residual)
  unsigned short* hb    = (unsigned short*)take((size_t)kBN * kFF * 2);
  unsigned short* attnb = qb;                       // reuse: q dead after logits
  unsigned short* x1b   = kb;                       // reuse: k' dead after logits
  float*          ffn   = proj;                     // reuse: proj dead after LN1

  const dim3 blk(256);
  const int nx = (int)(kBN * kH);                   // 25,165,824

  // 1. casts
  cast_f32_bf16<<<(nx + 255) / 256, blk, 0, stream>>>(x, xb, nx);
  cast_f32_bf16<<<(kH * kH + 255) / 256, blk, 0, stream>>>(Wq, wqb, kH * kH);
  cast_f32_bf16<<<(kH * kH + 255) / 256, blk, 0, stream>>>(Wk, wkb, kH * kH);
  cast_f32_bf16<<<(kH * kH + 255) / 256, blk, 0, stream>>>(Wv, wvb, kH * kH);
  cast_f32_bf16<<<(kH * kH + 255) / 256, blk, 0, stream>>>(Wo, wob, kH * kH);
  cast_f32_bf16<<<(kFF * kH + 255) / 256, blk, 0, stream>>>(fc1_w, fc1wb, kFF * kH);
  cast_f32_bf16<<<(kH * kFF + 255) / 256, blk, 0, stream>>>(fc2_w, fc2wb, kH * kFF);

  // 2. QKV projections (bf16 out), 128x128 tiles
  const dim3 gProj(kH / 128, (int)(kBN / 128), 1);
  gemm_bf16_kernel<4><<<gProj, blk, 0, stream>>>(xb, kH, wqb, kH, qb, nullptr, kH, nullptr, kH, EPI_BF16);
  gemm_bf16_kernel<4><<<gProj, blk, 0, stream>>>(xb, kH, wkb, kH, kb, nullptr, kH, nullptr, kH, EPI_BF16);
  gemm_bf16_kernel<4><<<gProj, blk, 0, stream>>>(xb, kH, wvb, kH, vb, nullptr, kH, nullptr, kH, EPI_BF16);

  // 3. k' = k + pe
  add_pe_kernel<<<nx / 256, blk, 0, stream>>>(kb, key_pe);

  // 4. logits = q^T k'  (per b,head)  -> 5. softmax -> probs (bf16)
  attn_logits_kernel<<<kB * kNH, blk, 0, stream>>>(qb, kb, logit);
  softmax_kernel<<<kB * kNH, blk, 0, stream>>>(logit, probs);

  // 6. out = v @ attn^T  (per-head batched GEMM, Nc=64 -> 128x64 tiles)
  gemm_bf16_kernel<2><<<dim3(1, kN / 128, kB * kNH), blk, 0, stream>>>(
      vb, kH, probs, kDH, attnb, nullptr, kH, nullptr, kDH, EPI_BF16 | EPI_HEADS);

  // 7. Wo projection (f32 out) -> 8. LN1 (x + proj)
  gemm_bf16_kernel<4><<<gProj, blk, 0, stream>>>(attnb, kH, wob, kH, nullptr, proj, kH, nullptr, kH, EPI_F32);
  ln_residual_kernel<<<(int)kBN, blk, 0, stream>>>(x, proj, ln1_g, ln1_b, x1, x1b);

  // 9. fc1 + bias + relu (bf16 out)
  gemm_bf16_kernel<4><<<dim3(kFF / 128, (int)(kBN / 128), 1), blk, 0, stream>>>(
      x1b, kH, fc1wb, kH, hb, nullptr, kFF, fc1_b, kH, EPI_BF16 | EPI_BIAS | EPI_RELU);

  // 10. fc2 + bias (f32 out) -> 11. LN3 (x1 + ffn) -> d_out
  gemm_bf16_kernel<4><<<gProj, blk, 0, stream>>>(
      hb, kFF, fc2wb, kFF, nullptr, ffn, kH, fc2_b, kFF, EPI_F32 | EPI_BIAS);
  ln_residual_kernel<<<(int)kBN, blk, 0, stream>>>(x1, ffn, ln3_g, ln3_b, (float*)d_out, nullptr);
}
